// NeuralLSLv2_18975165514075
// MI455X (gfx1250) — matrix operand order repaired and checked by reading, not compiled, not run on hardware
//
#include <hip/hip_runtime.h>
#include <hip/hip_bf16.h>
#include <math.h>
#include <stdint.h>

// ---------------------------------------------------------------------------
// NeuralLSL fused kernel for MI455X (gfx1250, wave32, WMMA + async LDS copy)
//
// Structure exploitation: qin separates into pos-only and batch-only parts:
//   attn_logits[b,p,k] = L0[p,k] + Lb[b,k]        (L0 precomputed once)
//   validity[b,p]      = Wv2 . gelu(Vpe[p]+Vse[b]) + bv2
// Per-batch GEMMs (M=64-row tiles, f16 WMMA, f32 accumulate):
//   h1..h3, logits, shift_embed, Qse, Vse.  Lb is a 64x64 K=256 VALU dot.
// Weights are pre-packed to f16 in the exact col-major chunk layout the
// B-fragments consume, so per-block staging is a contiguous async
// global->LDS b128 stream (ASYNCcnt) instead of load/convert/store.
// ---------------------------------------------------------------------------

#ifndef NLS_ASYNC
#define NLS_ASYNC 1          // flip to 0 to fall back to sync b128 copies
#endif

typedef _Float16 v16h __attribute__((ext_vector_type(16)));
typedef _Float16 v8h  __attribute__((ext_vector_type(8)));
typedef float    v8f  __attribute__((ext_vector_type(8)));

union AU { v16h v; v8h h[2]; };

#define DM   256   // d_model
#define NB   64    // BITS
#define ROWS 64    // batch rows per block
#define NT   256   // threads per block (8 wave32)

__device__ __forceinline__ float gelu_f(float x) {
  return 0.5f * x * (1.0f + erff(x * 0.70710678118654752f));
}
__device__ __forceinline__ v8f vzero() {
  v8f z = {0.f,0.f,0.f,0.f,0.f,0.f,0.f,0.f};
  return z;
}

// LDS byte offset of a generic pointer into this block's LDS allocation.
__device__ __forceinline__ uint32_t lds_off_u32(const void* p) {
  return (uint32_t)(uintptr_t)(const __attribute__((address_space(3))) void*)p;
}

#if NLS_ASYNC
// One 16B lane-copy: LDS[lds_byte] = MEM[s_base + v_off]  (ASYNCcnt-tracked)
__device__ __forceinline__ void async_cp_b128(uint32_t lds_byte, uint32_t goff,
                                              uint64_t gbase) {
  asm volatile("global_load_async_to_lds_b128 %0, %1, %2"
               :: "v"(lds_byte), "v"(goff), "s"(gbase) : "memory");
}
__device__ __forceinline__ void async_wait0() {
#if __has_builtin(__builtin_amdgcn_s_wait_asynccnt)
  __builtin_amdgcn_s_wait_asynccnt(0);
#else
  asm volatile("s_wait_asynccnt 0x0" ::: "memory");
#endif
}
#endif

// ---------------------------------------------------------------------------
// Pack one weight operand to f16 in B-fragment staging order:
//   dst[ch*(KD*64) + c*KD + k] = W[(rowoff+k)*ldw + ch*64 + c]
// ---------------------------------------------------------------------------
__global__ __launch_bounds__(256) void nls_pack(const float* __restrict__ W,
                                                _Float16* __restrict__ dst,
                                                int KD, int nch, int ldw,
                                                int rowoff)
{
  const int total = KD * 64 * nch;
  for (int e = blockIdx.x * blockDim.x + threadIdx.x; e < total;
       e += gridDim.x * blockDim.x) {
    const int ch = e / (KD * 64);
    const int rr = e - ch * (KD * 64);
    const int c  = rr / KD;
    const int k  = rr - c * KD;
    dst[e] = (_Float16)W[(size_t)(rowoff + k) * ldw + ch * 64 + c];
  }
}

// ---------------------------------------------------------------------------
// Kernel 0: batch-independent precompute (one block).
//   keys = pos@Wk+bk, qpe = pos@Wq_top+bq, vpe = pos@Wv1_top+bv1,
//   L0[p,k] = qpe[p].keys[k]/16
// ---------------------------------------------------------------------------
__global__ __launch_bounds__(256) void nls_precompute(
    const float* __restrict__ pos,
    const float* __restrict__ Wq, const float* __restrict__ bq,
    const float* __restrict__ Wk, const float* __restrict__ bk,
    const float* __restrict__ Wv1, const float* __restrict__ bv1,
    float* __restrict__ keys, float* __restrict__ qpe,
    float* __restrict__ vpe,  float* __restrict__ L0)
{
  const int t = threadIdx.x;
  for (int idx = t; idx < 64 * 256; idx += 256) {
    const int p = idx >> 8, d = idx & 255;
    float ak = bk[d], aq = bq[d], av = bv1[d];
    for (int j = 0; j < 256; ++j) {
      const float pe = pos[p * 256 + j];
      ak += pe * Wk [j * 256 + d];
      aq += pe * Wq [j * 256 + d];
      av += pe * Wv1[j * 256 + d];
    }
    keys[idx] = ak; qpe[idx] = aq; vpe[idx] = av;
  }
  __syncthreads();
  for (int idx = t; idx < 64 * 64; idx += 256) {
    const int p = idx >> 6, k = idx & 63;
    float a = 0.f;
    for (int d = 0; d < 256; ++d) a += qpe[p * 256 + d] * keys[k * 256 + d];
    L0[idx] = a * (1.0f / 16.0f);
  }
}

// ---------------------------------------------------------------------------
// Wave-cooperative GEMM: M=64 rows (f16 in lds_a, row-major, stride KD),
// N = 64*NCH cols in 64-col chunks; pre-packed f16 weights streamed into
// lds_b with async global->LDS b128. Wave w: mtile=w>>1, ntiles (w&1)*2+{0,1}.
// C kept in registers across chunks. Fragment layouts per ISA 7.12.2.
// ---------------------------------------------------------------------------
template<int KD, int NCH>
__device__ __forceinline__ void gemm(const _Float16* __restrict__ A,
                                     _Float16* __restrict__ Bl,
                                     const _Float16* __restrict__ Wh,
                                     v8f* C)
{
  const int t    = threadIdx.x;
  const int lane = t & 31, wave = t >> 5;
  const int hi   = lane >> 4;
  const int m    = (wave >> 1) * 16 + (lane & 15);
  for (int ch = 0; ch < NCH; ++ch) {
    __syncthreads();
#if NLS_ASYNC
    {
      const uint32_t lbase = lds_off_u32(Bl);
      const uint64_t gbase = (uint64_t)(uintptr_t)(Wh + (size_t)ch * KD * 64);
      for (int e = t * 16; e < KD * 128; e += NT * 16)
        async_cp_b128(lbase + (uint32_t)e, (uint32_t)e, gbase);
      async_wait0();
    }
#else
    for (int e = t * 8; e < KD * 64; e += NT * 8)
      *(v8h*)&Bl[e] = *(const v8h*)&Wh[(size_t)ch * KD * 64 + e];
#endif
    __syncthreads();
#pragma unroll
    for (int kk = 0; kk < KD; kk += 32) {
      AU a;  // A 16x32 f16 frag: lanes 0-15 K=kk+0..7/16..23, lanes 16-31 +8
      a.h[0] = *(const v8h*)&A[m * KD + kk + hi * 8];
      a.h[1] = *(const v8h*)&A[m * KD + kk + 16 + hi * 8];
#pragma unroll
      for (int j = 0; j < 2; ++j) {
        const int n = ((wave & 1) * 2 + j) * 16 + (lane & 15);
        AU b;  // B 32x16 f16 frag: lane=col, lanes 0-15 K=kk+0..7/16..23, lanes 16-31 +8
        b.h[0] = *(const v8h*)&Bl[n * KD + kk + hi * 8];
        b.h[1] = *(const v8h*)&Bl[n * KD + kk + 16 + hi * 8];
        C[ch * 2 + j] = __builtin_amdgcn_wmma_f32_16x16x32_f16(
            false, a.v, false, b.v, (short)0, C[ch * 2 + j], false, false);
      }
    }
  }
  __syncthreads();
}

// C/D layout: lanes 0-15 N=lane M=e, lanes 16-31 N=lane-16 M=e+8.
template<int NCH>
__device__ __forceinline__ void wb_f16(const v8f* C, const float* bias,
                                       _Float16* dst /* [64][256] */)
{
  const int t = threadIdx.x, lane = t & 31, wave = t >> 5, hi = lane >> 4;
  const int rbase = (wave >> 1) * 16 + hi * 8;
  for (int ch = 0; ch < NCH; ++ch)
#pragma unroll
    for (int j = 0; j < 2; ++j) {
      const int n = ch * 64 + ((wave & 1) * 2 + j) * 16 + (lane & 15);
      const float b = bias ? bias[n] : 0.0f;
#pragma unroll
      for (int e = 0; e < 8; ++e)
        dst[(rbase + e) * DM + n] = (_Float16)(C[ch * 2 + j][e] + b);
    }
}

// ---------------------------------------------------------------------------
// Main fused kernel: one block = 64 batch rows.
// LDS: lds_a 32KB (f16 activations / Vse), lds_b 32KB (weight chunks /
// f32 scratch: LN stats, logits, Lb+L0, Qse f16). Total exactly 64KB.
// ---------------------------------------------------------------------------
__global__ __launch_bounds__(256) void nls_main(
    const float* __restrict__ input_bits, const float* __restrict__ shift_bits,
    const float* __restrict__ b1, const float* __restrict__ g1, const float* __restrict__ be1,
    const float* __restrict__ b2, const float* __restrict__ g2, const float* __restrict__ be2,
    const float* __restrict__ b3, const float* __restrict__ g3, const float* __restrict__ be3,
    const float* __restrict__ b4,
    const float* __restrict__ Wv2, const float* __restrict__ bv2,
    const float* __restrict__ keys, const float* __restrict__ vpe,
    const float* __restrict__ L0g,
    const _Float16* __restrict__ W1h, const _Float16* __restrict__ W2h,
    const _Float16* __restrict__ W3h, const _Float16* __restrict__ W4h,
    const _Float16* __restrict__ posh, const _Float16* __restrict__ Wqh,
    const _Float16* __restrict__ Wv1h,
    float* __restrict__ out)
{
  __shared__ __align__(16) _Float16 lds_a[ROWS * DM];  // 32 KB
  __shared__ __align__(16) _Float16 lds_b[64 * DM];    // 32 KB
  float* lds_bf = reinterpret_cast<float*>(lds_b);

  const int t = threadIdx.x;
  const int lane = t & 31, wave = t >> 5, hi = lane >> 4;
  const int rowbase = blockIdx.x * ROWS;

  // LN(g,be) + exact-erf GELU over lds_a rows; stats via lds_b scratch.
  auto ln_gelu = [&](const float* g, const float* be) {
    const int r = t >> 2, q = t & 3;
    float s = 0.f, s2 = 0.f;
    for (int c = 0; c < 64; ++c) {
      const float x = (float)lds_a[r * DM + q * 64 + c];
      s += x; s2 += x * x;
    }
    lds_bf[r * 8 + q] = s; lds_bf[r * 8 + 4 + q] = s2;
    __syncthreads();
    float S = 0.f, S2 = 0.f;
    for (int qq = 0; qq < 4; ++qq) { S += lds_bf[r * 8 + qq]; S2 += lds_bf[r * 8 + 4 + qq]; }
    const float mean = S * (1.0f / DM);
    const float var  = S2 * (1.0f / DM) - mean * mean;
    const float inv  = rsqrtf(var + 1e-5f);
    for (int c = 0; c < 64; ++c) {
      const int cc = q * 64 + c;
      const float x = (float)lds_a[r * DM + cc];
      const float y = (x - mean) * inv * g[cc] + be[cc];
      lds_a[r * DM + cc] = (_Float16)gelu_f(y);
    }
    __syncthreads();
  };

  // ---- stage shift_bits tile (f32 -> f16, row-major stride 64) ----
  __builtin_prefetch(&input_bits[(size_t)rowbase * NB], 0, 3); // for epilogue
  for (int e = t; e < ROWS * NB; e += NT) {
    const int r = e >> 6, c = e & 63;
    lds_a[r * NB + c] = (_Float16)shift_bits[(size_t)(rowbase + r) * NB + c];
  }

  v8f Cm[8], Cq[8];

  // ---- MLP layer 1: [64,64] @ W1[64,256] ----
  for (int i = 0; i < 8; ++i) Cm[i] = vzero();
  gemm<NB, 4>(lds_a, lds_b, W1h, Cm);
  wb_f16<4>(Cm, b1, lds_a);  __syncthreads();  ln_gelu(g1, be1);

  // ---- MLP layer 2 ----
  for (int i = 0; i < 8; ++i) Cm[i] = vzero();
  gemm<DM, 4>(lds_a, lds_b, W2h, Cm);
  wb_f16<4>(Cm, b2, lds_a);  __syncthreads();  ln_gelu(g2, be2);

  // ---- MLP layer 3 ----
  for (int i = 0; i < 8; ++i) Cm[i] = vzero();
  gemm<DM, 4>(lds_a, lds_b, W3h, Cm);
  wb_f16<4>(Cm, b3, lds_a);  __syncthreads();  ln_gelu(g3, be3);

  // ---- shift logits: [64,256] @ W4[256,64] -> f32 in lds_bf ----
  Cm[0] = vzero(); Cm[1] = vzero();
  gemm<DM, 1>(lds_a, lds_b, W4h, Cm);
  {
    const int rbase = (wave >> 1) * 16 + hi * 8;
#pragma unroll
    for (int j = 0; j < 2; ++j) {
      const int n = ((wave & 1) * 2 + j) * 16 + (lane & 15);
      const float b = b4[n];
#pragma unroll
      for (int e = 0; e < 8; ++e)
        lds_bf[(rbase + e) * NB + n] = Cm[j][e] + b;
    }
  }
  __syncthreads();

  // ---- row softmax -> probs f16 into lds_a (stride 64) ----
  if (t < 64) {
    const float* lg = lds_bf + t * NB;
    float mx = -1e30f;
    for (int k = 0; k < NB; ++k) mx = fmaxf(mx, lg[k]);
    float den = 0.f;
    for (int k = 0; k < NB; ++k) den += __expf(lg[k] - mx);
    const float inv = 1.0f / den;
    for (int k = 0; k < NB; ++k)
      lds_a[t * NB + k] = (_Float16)(__expf(lg[k] - mx) * inv);
  }
  __syncthreads();

  // ---- shift_embed = probs[64,64] @ pos_embed[64,256] ----
  for (int i = 0; i < 8; ++i) Cm[i] = vzero();
  gemm<NB, 4>(lds_a, lds_b, posh, Cm);
  wb_f16<4>(Cm, nullptr, lds_a);  // se -> lds_a
  __syncthreads();

  // ---- Qse = se @ Wq[256:512], Vse = se @ Wv1[256:512] ----
  for (int i = 0; i < 8; ++i) Cq[i] = vzero();
  gemm<DM, 4>(lds_a, lds_b, Wqh, Cq);
  for (int i = 0; i < 8; ++i) Cm[i] = vzero();
  gemm<DM, 4>(lds_a, lds_b, Wv1h, Cm);
  wb_f16<4>(Cm, nullptr, lds_a);   // Vse f16 (bv1 folded into vpe)
  wb_f16<4>(Cq, nullptr, lds_b);   // Qse f16 -> lds_b
  __syncthreads();

  // ---- Lb[r,k] = Qse[r] . keys[k] / 16  (fixed k per thread, 16 rows) ----
  {
    const int kp = t & 63, rrb = t >> 6;
    float acc[16];
#pragma unroll
    for (int i = 0; i < 16; ++i) acc[i] = 0.f;
    const float* kr = keys + kp * DM;
    for (int d = 0; d < DM; ++d) {
      const float kv = kr[d];
#pragma unroll
      for (int i = 0; i < 16; ++i)
        acc[i] += kv * (float)lds_b[(i * 4 + rrb) * DM + d];
    }
    __syncthreads();  // done reading Qse
#pragma unroll
    for (int i = 0; i < 16; ++i)
      lds_bf[(i * 4 + rrb) * NB + kp] = acc[i] * 0.0625f;   // Lb at lds_bf[0..4095]
    for (int e = t; e < 64 * 64; e += NT)
      lds_bf[4096 + e] = L0g[e];                            // L0 at lds_bf[4096..]
  }
  __syncthreads();

  // ---- epilogue: per (r,p) attention softmax + gather + validity gate ----
  for (int i = 0; i < 16; ++i) {
    const int idx = i * NT + t;
    const int r = idx >> 6, p = idx & 63;
    const float* l0 = lds_bf + 4096 + p * NB;
    const float* lb = lds_bf + r * NB;
    float mx = -1e30f;
    for (int k = 0; k < NB; ++k) mx = fmaxf(mx, l0[k] + lb[k]);
    const float* ib = input_bits + (size_t)(rowbase + r) * NB;
    float den = 0.f, num = 0.f;
    for (int k = 0; k < NB; ++k) {
      const float e = __expf(l0[k] + lb[k] - mx);
      den += e; num += e * ib[k];
    }
    const float gathered = num / den;

    float v = bv2[0];
    const float* vp = vpe + p * DM;
    const _Float16* vs = lds_a + r * DM;
    for (int c = 0; c < DM; ++c) {
      const float x = vp[c] + (float)vs[c];
      v += gelu_f(x) * Wv2[c];
    }
    out[(size_t)(rowbase + r) * NB + p] = gathered * (1.0f / (1.0f + __expf(-v)));
  }
}

// ---------------------------------------------------------------------------
extern "C" void kernel_launch(void* const* d_in, const int* in_sizes, int n_in,
                              void* d_out, int out_size, void* d_ws, size_t ws_size,
                              hipStream_t stream) {
  const float* input_bits = (const float*)d_in[0];
  const float* shift_bits = (const float*)d_in[1];
  const float* W1  = (const float*)d_in[2];  const float* b1  = (const float*)d_in[3];
  const float* g1  = (const float*)d_in[4];  const float* be1 = (const float*)d_in[5];
  const float* W2  = (const float*)d_in[6];  const float* b2  = (const float*)d_in[7];
  const float* g2  = (const float*)d_in[8];  const float* be2 = (const float*)d_in[9];
  const float* W3  = (const float*)d_in[10]; const float* b3  = (const float*)d_in[11];
  const float* g3  = (const float*)d_in[12]; const float* be3 = (const float*)d_in[13];
  const float* W4  = (const float*)d_in[14]; const float* b4  = (const float*)d_in[15];
  const float* pos = (const float*)d_in[16];
  const float* Wq  = (const float*)d_in[17]; const float* bq  = (const float*)d_in[18];
  const float* Wk  = (const float*)d_in[19]; const float* bk  = (const float*)d_in[20];
  const float* Wv1 = (const float*)d_in[21]; const float* bv1 = (const float*)d_in[22];
  const float* Wv2 = (const float*)d_in[23]; const float* bv2 = (const float*)d_in[24];

  // f32 workspace
  float* ws   = (float*)d_ws;
  float* keys = ws;                 // 64*256
  float* qpe  = ws + 16384;         // 64*256 (precompute scratch)
  float* vpe  = ws + 32768;         // 64*256
  float* L0   = ws + 49152;         // 64*64
  // f16 packed weights (B-fragment staging order), 16B-aligned
  _Float16* wh   = (_Float16*)(ws + 53248);
  _Float16* W1h  = wh;              // 64*256   (KD=64,  NCH=4)
  _Float16* W2h  = W1h + 16384;     // 256*256  (KD=256, NCH=4)
  _Float16* W3h  = W2h + 65536;
  _Float16* W4h  = W3h + 65536;     // 256*64   (KD=256, NCH=1)
  _Float16* posh = W4h + 16384;     // 64*256   (KD=64,  NCH=4)
  _Float16* Wqh  = posh + 16384;    // 256*256  rows 256..511
  _Float16* Wv1h = Wqh + 65536;     // 256*256  rows 256..511

  nls_pack<<<16, 256, 0, stream>>>(W1,  W1h,  64, 4, 256, 0);
  nls_pack<<<64, 256, 0, stream>>>(W2,  W2h, 256, 4, 256, 0);
  nls_pack<<<64, 256, 0, stream>>>(W3,  W3h, 256, 4, 256, 0);
  nls_pack<<<16, 256, 0, stream>>>(W4,  W4h, 256, 1,  64, 0);
  nls_pack<<<16, 256, 0, stream>>>(pos, posh, 64, 4, 256, 0);
  nls_pack<<<64, 256, 0, stream>>>(Wq,  Wqh, 256, 4, 256, 256);
  nls_pack<<<64, 256, 0, stream>>>(Wv1, Wv1h, 256, 4, 256, 256);

  nls_precompute<<<1, 256, 0, stream>>>(pos, Wq, bq, Wk, bk, Wv1, bv1,
                                        keys, qpe, vpe, L0);

  const int B = in_sizes[0] / NB;   // 4096
  nls_main<<<B / ROWS, NT, 0, stream>>>(
      input_bits, shift_bits,
      b1, g1, be1, b2, g2, be2, b3, g3, be3, b4,
      Wv2, bv2, keys, vpe, L0,
      W1h, W2h, W3h, W4h, posh, Wqh, Wv1h,
      (float*)d_out);
}
